// HLinearAttention_11940009083668
// MI455X (gfx1250) — matrix operand order repaired and checked by reading, not compiled, not run on hardware
//
#include <hip/hip_runtime.h>
#include <stdint.h>

// ---------------------------------------------------------------------------
// CDNA5 / gfx1250 types for WMMA
// ---------------------------------------------------------------------------
typedef __bf16  bf16_t;
typedef __bf16  v16bf __attribute__((ext_vector_type(16)));
typedef __bf16  v8bf  __attribute__((ext_vector_type(8)));   // 16 B packed store
typedef __bf16  v4bf  __attribute__((ext_vector_type(4)));   //  8 B packed store
typedef float   v8f   __attribute__((ext_vector_type(8)));
typedef unsigned int v8u __attribute__((ext_vector_type(8)));

static __device__ __forceinline__ v8f wmma_bf16(v16bf a, v16bf b, v8f c) {
    // v_wmma_f32_16x16x32_bf16  (neg_a, A, neg_b, B, c_mod, C, reuse_a, reuse_b)
    return __builtin_amdgcn_wmma_f32_16x16x32_bf16(false, a, false, b, (short)0, c, false, false);
}

// A-matrix fragment (16x32 bf16, MxK). tile -> element [m=0][k=0], ldh = row
// stride in halves. Lane L holds row m = L&15; K placement per ISA 7.12.2:
//   lanes 0-15 : v0..v3 -> K 0..7,  v4..v7 -> K 16..23
//   lanes 16-31: v0..v3 -> K 8..15, v4..v7 -> K 24..31
static __device__ __forceinline__ v16bf ldA(const bf16_t* tile, int ldh, int lane) {
    const int m  = lane & 15;
    const int hi = lane >> 4;
    const bf16_t* row = tile + (size_t)m * ldh;
    v8u r;
#pragma unroll
    for (int j = 0; j < 8; ++j) {
        const int kb = (j < 4) ? (hi * 8 + 2 * j) : (16 + hi * 8 + 2 * (j - 4));
        r[j] = *(const unsigned int*)(row + kb);
    }
    return __builtin_bit_cast(v16bf, r);
}

// B-matrix fragment (32x16 bf16, KxN). tile -> element [k=0][n=0], ldh = row
// stride in halves. Lane L holds column N = L&15; K placement matches ldA.
static __device__ __forceinline__ v16bf ldB(const bf16_t* tile, int ldh, int lane) {
    const int n  = lane & 15;
    const int hi = lane >> 4;
    v16bf r;
#pragma unroll
    for (int s = 0; s < 16; ++s) {
        const int k = (s < 8) ? (hi * 8 + s) : (16 + hi * 8 + (s - 8));
        r[s] = tile[(size_t)k * ldh + n];
    }
    return r;
}

// pack 8 fp32 (two float4) -> one 16-byte bf16x8
static __device__ __forceinline__ v8bf pack8(float4 f0, float4 f1) {
    v8bf p;
    p[0] = (bf16_t)f0.x; p[1] = (bf16_t)f0.y; p[2] = (bf16_t)f0.z; p[3] = (bf16_t)f0.w;
    p[4] = (bf16_t)f1.x; p[5] = (bf16_t)f1.y; p[6] = (bf16_t)f1.z; p[7] = (bf16_t)f1.w;
    return p;
}

// ---------------------------------------------------------------------------
// Kernel 1: Out[M,N] = X[M,K] * W[N,K]^T + bias[N]     (fp32 in, bf16 WMMA)
// Block tile 64x256xBK64, 8 waves (2x4), each wave 32x64 (2x4 of 16x16 tiles).
// M=4096, K=2048, N in {2048, 512} -- all tile-divisible, no guards needed.
// ---------------------------------------------------------------------------
__global__ __launch_bounds__(256)
void gla_proj_gemm(const float* __restrict__ X, const float* __restrict__ W,
                   const float* __restrict__ bias, float* __restrict__ Out,
                   int N, int K) {
    __shared__ bf16_t Xs[64][64];    //  8 KB  (m-major, k contiguous)
    __shared__ bf16_t Ws[64][256];   // 32 KB  (k-major, n contiguous)

    const int tid  = threadIdx.x;
    const int lane = tid & 31;
    const int w    = tid >> 5;       // 0..7
    const int wm   = w & 1;          // 2 wave-rows
    const int wn   = w >> 1;         // 4 wave-cols
    const int m0   = blockIdx.x * 64;
    const int n0   = blockIdx.y * 256;

    // X staging: thread owns 16 contiguous k at one row
    const int xrow = tid >> 2;             // 0..63
    const int xcg  = (tid & 3) * 16;       // 0,16,32,48
    // W staging: thread owns 4 consecutive n-rows x 16 contiguous k
    const int wn4  = (tid & 63) * 4;       // 0..252
    const int wkg  = (tid >> 6) * 16;      // 0,16,32,48

    v8f acc[2][4];
    const v8f vz = {};
#pragma unroll
    for (int i = 0; i < 2; ++i)
#pragma unroll
        for (int j = 0; j < 4; ++j) acc[i][j] = vz;

    for (int kk = 0; kk < K; kk += 64) {
        // ---- stage X tile 64x64: 4x float4 loads -> 2x b128 LDS stores
        {
            const float4* src = (const float4*)(X + (size_t)(m0 + xrow) * K + kk + xcg);
            float4 f0 = src[0], f1 = src[1], f2 = src[2], f3 = src[3];
            *(v8bf*)&Xs[xrow][xcg]     = pack8(f0, f1);
            *(v8bf*)&Xs[xrow][xcg + 8] = pack8(f2, f3);
        }
        // ---- stage W tile 256x64 transposed: 4n x 16k micro-tile per thread,
        //      packed 4-halves-along-n -> 16x ds_store_b64
        {
            float wr[4][16];
#pragma unroll
            for (int rr = 0; rr < 4; ++rr) {
                const float4* src = (const float4*)(W + (size_t)(n0 + wn4 + rr) * K + kk + wkg);
#pragma unroll
                for (int q = 0; q < 4; ++q) {
                    float4 t = src[q];
                    wr[rr][q * 4 + 0] = t.x; wr[rr][q * 4 + 1] = t.y;
                    wr[rr][q * 4 + 2] = t.z; wr[rr][q * 4 + 3] = t.w;
                }
            }
#pragma unroll
            for (int k = 0; k < 16; ++k) {
                v4bf p;
                p[0] = (bf16_t)wr[0][k]; p[1] = (bf16_t)wr[1][k];
                p[2] = (bf16_t)wr[2][k]; p[3] = (bf16_t)wr[3][k];
                *(v4bf*)&Ws[wkg + k][wn4] = p;
            }
        }
        // prefetch next K tile while this one is consumed
        if (kk + 64 < K) {
            __builtin_prefetch(X + (size_t)(m0 + xrow) * K + kk + 64 + xcg, 0, 0);
            __builtin_prefetch(W + (size_t)(n0 + wn4) * K + kk + 64 + wkg, 0, 0);
        }
        __syncthreads();

        // ---- 16 WMMAs per stage (2 k-substeps x 2m x 4n)
#pragma unroll
        for (int ks = 0; ks < 2; ++ks) {
            const v16bf a0 = ldA(&Xs[wm * 32][ks * 32],      64, lane);
            const v16bf a1 = ldA(&Xs[wm * 32 + 16][ks * 32], 64, lane);
#pragma unroll
            for (int ni = 0; ni < 4; ++ni) {
                const v16bf bfr = ldB(&Ws[ks * 32][wn * 64 + ni * 16], 256, lane);
                acc[0][ni] = wmma_bf16(a0, bfr, acc[0][ni]);
                acc[1][ni] = wmma_bf16(a1, bfr, acc[1][ni]);
            }
        }
        __syncthreads();
    }

    const int hi = lane >> 4, nl = lane & 15;
#pragma unroll
    for (int mi = 0; mi < 2; ++mi)
#pragma unroll
        for (int ni = 0; ni < 4; ++ni) {
            const int n = n0 + wn * 64 + ni * 16 + nl;
            const float bv = bias[n];
#pragma unroll
            for (int r = 0; r < 8; ++r) {
                const int m = m0 + wm * 32 + mi * 16 + hi * 8 + r;
                Out[(size_t)m * N + n] = acc[mi][ni][r] + bv;
            }
        }
}

// ---------------------------------------------------------------------------
// Kernel 2: chunked GLA recurrence. One block (8 wave32) per (b,h).
// Chunk C=16 rows, D=128. State S[128][128] fp32 + bf16 mirror in LDS.
//   q~_t = relu(q_t)*scale*e^{A_t} ; k~_i = kf_i*e^{-A_i} ; k^_i = kf_i*e^{A_C-A_i}
//   O = q~ . S0  +  causal(q~ . k~^T) . V ;   S <- diag(e^{A_C}).S0 + k^T . V
// ---------------------------------------------------------------------------
#define GLA_SEQ   2048
#define GLA_DM    2048
#define GLA_KV    512
#define GLA_SMEM  124416

__global__ __launch_bounds__(256)
void gla_recurrence(const float* __restrict__ qp, const float* __restrict__ kp,
                    const float* __restrict__ vp, float* __restrict__ out) {
    extern __shared__ char smem[];
    float*  Sst = (float*) (smem);            // [128][128] fp32 state   64 KB
    bf16_t* Sbf = (bf16_t*)(smem +  65536);   // [128][128] bf16 mirror  32 KB
    bf16_t* Qt  = (bf16_t*)(smem +  98304);   // [16][128]  q~            4 KB
    bf16_t* KtT = (bf16_t*)(smem + 102400);   // [128][16]  k~^T          4 KB
    bf16_t* KhT = (bf16_t*)(smem + 106496);   // [128][32]  k^T (pad)     8 KB
    bf16_t* Vp  = (bf16_t*)(smem + 114688);   // [32][128]  V (pad)       8 KB
    bf16_t* Pm  = (bf16_t*)(smem + 122880);   // [16][32]   masked P      1 KB
    float*  dec = (float*) (smem + 123904);   // [128]      e^{A_C}      .5 KB

    const int tid  = threadIdx.x;
    const int lane = tid & 31;
    const int w    = tid >> 5;
    const int b    = blockIdx.x >> 4;
    const int h    = blockIdx.x & 15;
    const int hk   = h >> 2;                  // GQA repeat = 4
    const size_t row0 = (size_t)b * GLA_SEQ;  // row base into [B*S, *] arrays
    const float scale = 0.08838834764831845f; // 128^-0.5

    // init: zero state and the permanently-zero pad regions
    for (int i = tid; i < 128 * 128; i += 256) Sst[i] = 0.f;
    for (int i = tid; i < 16 * 128;  i += 256) Vp[16 * 128 + i] = (bf16_t)0.f;     // V rows 16..31
    for (int i = tid; i < 16 * 16;   i += 256) Pm[(i >> 4) * 32 + 16 + (i & 15)]  = (bf16_t)0.f;
    __syncthreads();

    for (int t0 = 0; t0 < GLA_SEQ; t0 += 16) {
        // --- phase 1a: mirror state to bf16 (64 contiguous elems per thread,
        //     packed b128 loads/stores)
        {
            const int base = tid * 64;
            const float4* s = (const float4*)(Sst + base);
#pragma unroll
            for (int jj = 0; jj < 8; ++jj)
                *(v8bf*)(Sbf + base + jj * 8) = pack8(s[jj * 2], s[jj * 2 + 1]);
        }
        // --- phase 1b: load V chunk (float4 loads, one b128 LDS store)
        {
            const int row = tid >> 4;
            const int cg  = (tid & 15) * 8;
            const float4* src = (const float4*)(vp + (row0 + t0 + row) * GLA_KV + hk * 128 + cg);
            *(v8bf*)&Vp[row * 128 + cg] = pack8(src[0], src[1]);
        }
        // --- phase 1c: per-feature gate cumsum; row-packed k~ / k^ stores
        if (tid < 128) {
            const int d = tid;
            float kf[16], Ac[16];
            float gcum = 0.f;
#pragma unroll
            for (int t = 0; t < 16; ++t) {
                const float kraw = kp[(row0 + t0 + t) * GLA_KV + hk * 128 + d];
                const float sig  = 1.f / (1.f + __expf(-kraw));
                const float kfv  = fminf(sig, 0.95f);
                gcum += log1pf(-kfv);
                kf[t] = kfv;  Ac[t] = gcum;
                float qv = qp[(row0 + t0 + t) * GLA_DM + h * 128 + d];
                qv = fmaxf(qv, 0.f) * scale;
                Qt[t * 128 + d] = (bf16_t)(qv * __expf(gcum));     // scatter (layout-forced)
            }
            const float Al = Ac[15];
            dec[d] = __expf(Al);
            v8bf kt0, kt1, kh0, kh1;
            const v8bf zv = {};
#pragma unroll
            for (int t = 0; t < 8; ++t) {
                kt0[t] = (bf16_t)(kf[t]     * __expf(-Ac[t]));      // bf16 exp range = fp32
                kt1[t] = (bf16_t)(kf[t + 8] * __expf(-Ac[t + 8]));
                kh0[t] = (bf16_t)(kf[t]     * __expf(Al - Ac[t]));
                kh1[t] = (bf16_t)(kf[t + 8] * __expf(Al - Ac[t + 8]));
            }
            *(v8bf*)&KtT[d * 16]     = kt0;  *(v8bf*)&KtT[d * 16 + 8]  = kt1;
            *(v8bf*)&KhT[d * 32]     = kh0;  *(v8bf*)&KhT[d * 32 + 8]  = kh1;
            *(v8bf*)&KhT[d * 32 + 16] = zv;  *(v8bf*)&KhT[d * 32 + 24] = zv;
            // prefetch next chunk's gate/query rows
            if (t0 + 16 < GLA_SEQ) {
                __builtin_prefetch(kp + (row0 + t0 + 16) * GLA_KV + hk * 128 + d, 0, 0);
                __builtin_prefetch(qp + (row0 + t0 + 16) * GLA_DM + h * 128 + d, 0, 0);
            }
        }
        __syncthreads();

        // --- phase 2a: inter-chunk O = q~ . S0 ; wave 0 also builds masked P
        const int n0 = w * 16;
        v8f acc = {};
#pragma unroll
        for (int ks = 0; ks < 4; ++ks) {
            const v16bf a   = ldA(Qt + ks * 32, 128, lane);
            const v16bf bfr = ldB(Sbf + (size_t)(ks * 32) * 128 + n0, 128, lane);
            acc = wmma_bf16(a, bfr, acc);
        }
        if (w == 0) {
            v8f pacc = {};
#pragma unroll
            for (int ks = 0; ks < 4; ++ks) {
                const v16bf a   = ldA(Qt + ks * 32, 128, lane);
                const v16bf bfr = ldB(KtT + (size_t)(ks * 32) * 16, 16, lane);
                pacc = wmma_bf16(a, bfr, pacc);
            }
            const int hi = lane >> 4, ic = lane & 15;
#pragma unroll
            for (int r = 0; r < 8; ++r) {
                const int tr = hi * 8 + r;
                Pm[tr * 32 + ic] = (bf16_t)((ic <= tr) ? pacc[r] : 0.f);
            }
        }
        __syncthreads();

        // --- phase 2b: intra-chunk O += causal(P) . V ; write O chunk
        {
            const v16bf a   = ldA(Pm, 32, lane);
            const v16bf bfr = ldB(Vp + n0, 128, lane);
            acc = wmma_bf16(a, bfr, acc);
        }
        {
            const int hi = lane >> 4;
            const int e  = n0 + (lane & 15);
#pragma unroll
            for (int r = 0; r < 8; ++r) {
                const int t = hi * 8 + r;
                out[(row0 + t0 + t) * GLA_DM + h * 128 + e] = acc[r];
            }
        }

        // --- phase 2c: state update  S = diag(dec).S0 + k^T . V
        {
            const int m0d = w * 16;
            const v16bf a = ldA(KhT + (size_t)m0d * 32, 32, lane);
            const int hi = lane >> 4, nl = lane & 15;
#pragma unroll
            for (int et = 0; et < 8; ++et) {
                v8f sa = {};
                const v16bf bfr = ldB(Vp + et * 16, 128, lane);
                sa = wmma_bf16(a, bfr, sa);
#pragma unroll
                for (int r = 0; r < 8; ++r) {
                    const int dd = m0d + hi * 8 + r;
                    const int e  = et * 16 + nl;
                    Sst[dd * 128 + e] = Sst[dd * 128 + e] * dec[dd] + sa[r];
                }
            }
        }
        __syncthreads();
    }
}

// ---------------------------------------------------------------------------
// Host-side launch
// ---------------------------------------------------------------------------
extern "C" void kernel_launch(void* const* d_in, const int* in_sizes, int n_in,
                              void* d_out, int out_size, void* d_ws, size_t ws_size,
                              hipStream_t stream) {
    (void)in_sizes; (void)n_in; (void)out_size; (void)ws_size;

    const float* X  = (const float*)d_in[0];   // [B,S,DM] = [4096, 2048]
    const float* Wq = (const float*)d_in[1];   // [2048, 2048]
    const float* bq = (const float*)d_in[2];
    const float* Wk = (const float*)d_in[3];   // [512, 2048]
    const float* bk = (const float*)d_in[4];
    const float* Wv = (const float*)d_in[5];   // [512, 2048]
    const float* bv = (const float*)d_in[6];
    float* out = (float*)d_out;                // [B,S,2048]

    const int M = 4096, K = 2048;
    float* qproj = (float*)d_ws;                          // 4096x2048
    float* kproj = qproj + (size_t)M * 2048;              // 4096x512
    float* vproj = kproj + (size_t)M * 512;               // 4096x512

    dim3 blk(256);
    gla_proj_gemm<<<dim3(M / 64, 2048 / 256), blk, 0, stream>>>(X, Wq, bq, qproj, 2048, K);
    gla_proj_gemm<<<dim3(M / 64,  512 / 256), blk, 0, stream>>>(X, Wk, bk, kproj,  512, K);
    gla_proj_gemm<<<dim3(M / 64,  512 / 256), blk, 0, stream>>>(X, Wv, bv, vproj,  512, K);

    gla_recurrence<<<dim3(32), blk, GLA_SMEM, stream>>>(qproj, kproj, vproj, out);
}
